// LSTMDecoder_33371895889944
// MI455X (gfx1250) — compile-verified, matrix operand order checked by reading, not run on hardware
//
#include <hip/hip_runtime.h>
#include <hip/hip_bf16.h>

// ---------------- problem constants ----------------
#define B_SZ 512
#define S_SZ 200
#define NCW  10
#define CW   6
#define D_M  1024
#define LD   256
#define P_SZ 256
#define DIN  1280          // D_M + LD
#define G4   4096          // 4*D_M
#define LN_EPS 1e-5f

// ---------------- workspace layout (bytes) ----------------
// zi   : f16  [200][4096][512]               = 838,860,800
// WihP : bf16 packed B-fragments (4096x1280) =  10,485,760
// WhhP : bf16 packed B-fragments (4096x256)  =   2,097,152
// WhrP : bf16 packed B-fragments (256x1024)  =     524,288
// hs   : f32  [512*200][256]                 = 104,857,600
static constexpr size_t OFF_ZI  = 0;
static constexpr size_t OFF_WIH = OFF_ZI  + (size_t)S_SZ*G4*B_SZ*2;
static constexpr size_t OFF_WHH = OFF_WIH + (size_t)G4*DIN*2;
static constexpr size_t OFF_WHR = OFF_WHH + (size_t)G4*P_SZ*2;
static constexpr size_t OFF_HS  = OFF_WHR + (size_t)P_SZ*D_M*2;

// ---------------- types ----------------
typedef __bf16 bf16;
typedef __attribute__((ext_vector_type(16))) __bf16        v16bf;
typedef __attribute__((ext_vector_type(8)))  float         v8f;
typedef __attribute__((ext_vector_type(8)))  unsigned int  v8u;
typedef __attribute__((ext_vector_type(4)))  unsigned int  v4u;
typedef __attribute__((ext_vector_type(8)))  _Float16      v8h;

// ---------------- helpers ----------------
__device__ __forceinline__ unsigned short f32_to_bf16_rne(float f) {
  unsigned int u = __builtin_bit_cast(unsigned int, f);
  unsigned int r = u + 0x7FFFu + ((u >> 16) & 1u);
  return (unsigned short)(r >> 16);
}
__device__ __forceinline__ unsigned int pack_bf16x2(float a, float b) {
  return (unsigned)f32_to_bf16_rne(a) | ((unsigned)f32_to_bf16_rne(b) << 16);
}
__device__ __forceinline__ float sigf(float x) { return 1.0f / (1.0f + __expf(-x)); }

__device__ __forceinline__ v8f wmma_bf16(v8u a, v8u b, v8f c) {
  return __builtin_amdgcn_wmma_f32_16x16x32_bf16(
      false, __builtin_bit_cast(v16bf, a),
      false, __builtin_bit_cast(v16bf, b),
      (short)0, c, false, false);
}

// A-fragment (16x32 bf16, M x K) gathered from row-major LDS (stride in elems, even).
// Lane layout per ISA 7.12.2: lanes 0-15 -> M=lane, K-pairs {0,2,4,6,16,18,20,22};
// lanes 16-31 -> M=lane-16, K-pairs {8,10,12,14,24,26,28,30}.
// Pairs are contiguous dwords -> compiler fuses to 2x ds_load_b128 per fragment.
__device__ __forceinline__ v8u lds_afrag(const bf16* sm, int stride, int k0) {
  const int lane = threadIdx.x & 31;
  const int m = lane & 15, hi = (lane >> 4) & 1;
  const unsigned int* row = (const unsigned int*)(sm + m * stride);
  v8u r;
#pragma unroll
  for (int v = 0; v < 8; ++v) {
    int k = k0 + 2 * v + ((v < 4) ? 0 : 8) + (hi ? 8 : 0);
    r[v] = row[k >> 1];
  }
  return r;
}

// B-fragment: pre-packed 32 lanes x 8 dwords per (n_tile,k_tile) tile -> 2x global_load_b128.
__device__ __forceinline__ v8u g_bfrag(const unsigned int* pk, int tile) {
  const int lane = threadIdx.x & 31;
  return *(const v8u*)(pk + ((size_t)tile << 8) + (lane << 3));
}

// ---------------- kernel: pack W (row-major [N][K], f32) into bf16 B-fragments ----------------
// B[k][n] = W[n][k]; out dword idx = ((n_tile*nK + k_tile)*32 + lane)*8 + v
__global__ void k_pack(const float* __restrict__ W, unsigned int* __restrict__ out,
                       int N, int K) {
  int idx = blockIdx.x * 256 + threadIdx.x;
  int total = (N >> 4) * (K >> 5) * 256;
  if (idx >= total) return;
  int v    = idx & 7;
  int lane = (idx >> 3) & 31;
  int tile = idx >> 8;
  int nK   = K >> 5;
  int kt = tile % nK, nt = tile / nK;
  int hi = lane >> 4;
  int n = nt * 16 + (lane & 15);
  int k = kt * 32 + 2 * v + ((v < 4) ? 0 : 8) + (hi ? 8 : 0);
  out[idx] = pack_bf16x2(W[(size_t)n * K + k], W[(size_t)n * K + k + 1]);
}

// ---------------- kernel 1: zi = [emb | ctx] @ W_ih^T + b_ih + b_hh, stored f16 [t][n][b] ----
// One block = full 4096-col slab for one (t, 16-row batch tile): emb computed once per slab.
__global__ __launch_bounds__(512) void k_zi(
    const float* __restrict__ x, const float* __restrict__ ctx,
    const float* __restrict__ Wcmd, const float* __restrict__ bcmd,
    const float* __restrict__ Wcrd, const float* __restrict__ bcrd,
    const float* __restrict__ bih, const float* __restrict__ bhh,
    const unsigned int* __restrict__ WihP, _Float16* __restrict__ zi) {
  __shared__ bf16  zl[16 * 1288];   // z tile, stride 1280+8 to avoid LDS bank conflicts
  __shared__ float xs[16 * 16];
  const int t   = blockIdx.y;
  const int b0  = blockIdx.x * 16;
  const int tid = threadIdx.x;

  if (tid < 256) {
    int m = tid >> 4, k = tid & 15;
    xs[tid] = x[((size_t)(b0 + m) * S_SZ + t) * 16 + k];
  }
  __syncthreads();
  // emb (rank-16 input projection) -> bf16 LDS
  for (int e = tid; e < 16 * D_M; e += 512) {
    int j = e >> 4, m = e & 15;
    float acc = bcmd[j] + bcrd[j];
    const float* wc = Wcmd + j * NCW;
    const float* wd = Wcrd + j * CW;
#pragma unroll
    for (int k = 0; k < NCW; ++k) acc += xs[m * 16 + k] * wc[k];
#pragma unroll
    for (int k = 0; k < CW; ++k)  acc += xs[m * 16 + NCW + k] * wd[k];
    zl[m * 1288 + j] = (bf16)acc;
  }
  for (int e = tid; e < 16 * LD; e += 512) {
    int j = e >> 4, m = e & 15;
    zl[m * 1288 + D_M + j] = (bf16)ctx[(size_t)(b0 + m) * LD + j];
  }
  __syncthreads();

  const int wave = tid >> 5, lane = tid & 31;
  v8f acc[16] = {};
  for (int kt = 0; kt < 40; ++kt) {
    v8u a = lds_afrag(zl, 1288, kt * 32);
#pragma unroll
    for (int j = 0; j < 16; ++j) {
      v8u b = g_bfrag(WihP, (wave * 16 + j) * 40 + kt);
      acc[j] = wmma_bf16(a, b, acc[j]);
    }
  }
  const int m0 = (lane >> 4) * 8;
#pragma unroll
  for (int j = 0; j < 16; ++j) {
    int n = (wave * 16 + j) * 16 + (lane & 15);
    float bv = bih[n] + bhh[n];
    v8h h;
#pragma unroll
    for (int i = 0; i < 8; ++i) h[i] = (_Float16)(acc[j][i] + bv);
    // [t][n][b] layout: 8 consecutive batch rows -> one 16B store
    *(v4u*)(zi + ((size_t)t * G4 + n) * B_SZ + b0 + m0) = __builtin_bit_cast(v4u, h);
  }
}

// ---------------- kernel 2: LSTM scan (per-batch-row recurrence, 16 rows / block) ----------
__global__ __launch_bounds__(512) void k_scan(
    const unsigned int* __restrict__ WhhP, const unsigned int* __restrict__ WhrP,
    const _Float16* __restrict__ zi, float* __restrict__ hs) {
  __shared__ bf16 hA[16 * 264];    // h     (16 x 256),  padded stride
  __shared__ bf16 hR[16 * 1032];   // h_raw (16 x 1024), padded stride
  const int tid  = threadIdx.x;
  const int wave = tid >> 5, lane = tid & 31;
  const int b0   = blockIdx.x * 16;
  const int m0   = (lane >> 4) * 8;
  const int nlo  = lane & 15;

  for (int e = tid; e < 16 * 264; e += 512) hA[e] = (bf16)0.0f;
  v8f cst[4] = {};   // cell state, 4 column-tiles per wave, kept in registers
  __syncthreads();

  for (int t = 0; t < S_SZ; ++t) {
    // Issue the zi_t stream first: these 16x16B loads are independent of LDS state and
    // overlap the entire 128-WMMA gates GEMM below. Prefetch t+1 into WGP$/L2.
    v4u zr[16];
#pragma unroll
    for (int j = 0; j < 16; ++j) {
      int n = (wave + 16 * j) * 16 + nlo;
      const _Float16* zp = zi + ((size_t)t * G4 + n) * B_SZ + b0 + m0;
      zr[j] = *(const v4u*)zp;
      if (t + 1 < S_SZ)
        __builtin_prefetch(zp + (size_t)G4 * B_SZ, 0, 1);   // global_prefetch_b8
    }
    // gates = h @ Whh^T (+ zi_t). Wave w owns n-tiles {w + 16j}: same lane holds i/f/g/o
    // of the same hidden index in acc[jj], acc[4+jj], acc[8+jj], acc[12+jj].
    v8f acc[16] = {};
    for (int kt = 0; kt < 8; ++kt) {
      v8u a = lds_afrag(hA, 264, kt * 32);
#pragma unroll
      for (int j = 0; j < 16; ++j) {
        v8u b = g_bfrag(WhhP, (wave + 16 * j) * 8 + kt);
        acc[j] = wmma_bf16(a, b, acc[j]);
      }
    }
#pragma unroll
    for (int j = 0; j < 16; ++j) {
      v8h z8 = __builtin_bit_cast(v8h, zr[j]);
#pragma unroll
      for (int i = 0; i < 8; ++i) acc[j][i] += (float)z8[i];
    }
    // activations + cell update (pure per-lane VALU), h_raw -> LDS bf16
#pragma unroll
    for (int jj = 0; jj < 4; ++jj) {
      int dcol = (wave + 16 * jj) * 16 + nlo;
#pragma unroll
      for (int i = 0; i < 8; ++i) {
        float iv = sigf(acc[jj][i]);
        float fv = sigf(acc[4 + jj][i]);
        float gv = tanhf(acc[8 + jj][i]);
        float ov = sigf(acc[12 + jj][i]);
        float cv = fv * cst[jj][i] + iv * gv;
        cst[jj][i] = cv;
        hR[(m0 + i) * 1032 + dcol] = (bf16)(ov * tanhf(cv));
      }
    }
    __syncthreads();
    // h = h_raw @ Whr^T : wave w -> output n-tile w, K = 1024
    v8f hacc = {};
    for (int kt = 0; kt < 32; ++kt) {
      v8u a = lds_afrag(hR, 1032, kt * 32);
      v8u b = g_bfrag(WhrP, wave * 32 + kt);
      hacc = wmma_bf16(a, b, hacc);
    }
    int n = wave * 16 + nlo;
#pragma unroll
    for (int i = 0; i < 8; ++i) {
      float hv = hacc[i];
      hA[(m0 + i) * 264 + n] = (bf16)hv;                                   // next-step A operand
      hs[((size_t)(b0 + m0 + i) * S_SZ + t) * P_SZ + n] = hv;              // sequence output
    }
    __syncthreads();
  }
}

// ---------------- kernel 3: LayerNorm + command/coord heads (one token per wave32) --------
__global__ __launch_bounds__(256) void k_head(
    const float* __restrict__ hs, const float* __restrict__ ln_g, const float* __restrict__ ln_b,
    const float* __restrict__ Woc, const float* __restrict__ boc,
    const float* __restrict__ Wox, const float* __restrict__ box,
    const float* __restrict__ cscale, float* __restrict__ out_cmd, float* __restrict__ out_crd) {
  const int wave = threadIdx.x >> 5, lane = threadIdx.x & 31;
  const size_t tk = (size_t)blockIdx.x * 8 + wave;
  const float* hrow = hs + tk * P_SZ + lane * 8;
  float v[8];
#pragma unroll
  for (int i = 0; i < 8; ++i) v[i] = hrow[i];
  float s = 0.f, ss = 0.f;
#pragma unroll
  for (int i = 0; i < 8; ++i) { s += v[i]; ss += v[i] * v[i]; }
#pragma unroll
  for (int m = 16; m >= 1; m >>= 1) { s += __shfl_xor(s, m, 32); ss += __shfl_xor(ss, m, 32); }
  float mu  = s * (1.f / 256.f);
  float var = ss * (1.f / 256.f) - mu * mu;
  float inv = rsqrtf(var + LN_EPS);
  float hn[8];
#pragma unroll
  for (int i = 0; i < 8; ++i) {
    int n = lane * 8 + i;
    hn[i] = (v[i] - mu) * inv * ln_g[n] + ln_b[n];
  }
  float lg[10];
#pragma unroll
  for (int j = 0; j < 10; ++j) {
    float p = 0.f;
    const float* w = Woc + j * P_SZ + lane * 8;
#pragma unroll
    for (int i = 0; i < 8; ++i) p += hn[i] * w[i];
#pragma unroll
    for (int m = 16; m >= 1; m >>= 1) p += __shfl_xor(p, m, 32);
    lg[j] = p + boc[j];
  }
  float sc = *cscale;
  float co[6];
#pragma unroll
  for (int j = 0; j < 6; ++j) {
    float p = 0.f;
    const float* w = Wox + j * 266 + lane * 8;
#pragma unroll
    for (int i = 0; i < 8; ++i) p += hn[i] * w[i];
#pragma unroll
    for (int m = 16; m >= 1; m >>= 1) p += __shfl_xor(p, m, 32);
    const float* wl = Wox + j * 266 + 256;
#pragma unroll
    for (int k = 0; k < 10; ++k) p += lg[k] * wl[k];
    co[j] = tanhf((p + box[j]) * sc);
  }
  if (lane == 0) {
#pragma unroll
    for (int j = 0; j < 10; ++j) out_cmd[tk * 10 + j] = lg[j];
#pragma unroll
    for (int j = 0; j < 6; ++j)  out_crd[tk * 6 + j]  = co[j];
  }
}

// ---------------- launcher ----------------
extern "C" void kernel_launch(void* const* d_in, const int* in_sizes, int n_in,
                              void* d_out, int out_size, void* d_ws, size_t ws_size,
                              hipStream_t stream) {
  (void)in_sizes; (void)n_in; (void)out_size; (void)ws_size;
  const float* x    = (const float*)d_in[0];
  const float* ctx  = (const float*)d_in[1];
  const float* Wcmd = (const float*)d_in[2];
  const float* bcmd = (const float*)d_in[3];
  const float* Wcrd = (const float*)d_in[4];
  const float* bcrd = (const float*)d_in[5];
  const float* Wih  = (const float*)d_in[6];
  const float* bih  = (const float*)d_in[7];
  const float* Whh  = (const float*)d_in[8];
  const float* bhh  = (const float*)d_in[9];
  const float* Whr  = (const float*)d_in[10];
  const float* lng  = (const float*)d_in[11];
  const float* lnb  = (const float*)d_in[12];
  const float* Woc  = (const float*)d_in[13];
  const float* boc  = (const float*)d_in[14];
  const float* Wox  = (const float*)d_in[15];
  const float* box  = (const float*)d_in[16];
  const float* csc  = (const float*)d_in[17];

  char* ws = (char*)d_ws;
  _Float16*     zi   = (_Float16*)(ws + OFF_ZI);
  unsigned int* WihP = (unsigned int*)(ws + OFF_WIH);
  unsigned int* WhhP = (unsigned int*)(ws + OFF_WHH);
  unsigned int* WhrP = (unsigned int*)(ws + OFF_WHR);
  float*        hs   = (float*)(ws + OFF_HS);

  // pack weights into WMMA B-fragment order (bf16)
  k_pack<<<dim3((G4 * DIN / 2 + 255) / 256), dim3(256), 0, stream>>>(Wih, WihP, G4, DIN);
  k_pack<<<dim3((G4 * P_SZ / 2 + 255) / 256), dim3(256), 0, stream>>>(Whh, WhhP, G4, P_SZ);
  k_pack<<<dim3((P_SZ * D_M / 2 + 255) / 256), dim3(256), 0, stream>>>(Whr, WhrP, P_SZ, D_M);

  // big input-projection GEMM (WMMA), zi stored f16 [t][n][b]
  k_zi<<<dim3(32, 200), dim3(512), 0, stream>>>(x, ctx, Wcmd, bcmd, Wcrd, bcrd,
                                                bih, bhh, WihP, zi);
  // recurrent scan: 32 blocks x 16 waves, weights L2-resident
  k_scan<<<dim3(32), dim3(512), 0, stream>>>(WhhP, WhrP, zi, hs);

  float* out_cmd = (float*)d_out;
  float* out_crd = out_cmd + (size_t)B_SZ * S_SZ * NCW;
  k_head<<<dim3(B_SZ * S_SZ / 8), dim3(256), 0, stream>>>(hs, lng, lnb, Woc, boc,
                                                          Wox, box, csc, out_cmd, out_crd);
}
